// SegmentGenerator_14070312862061
// MI455X (gfx1250) — matrix operand order repaired and checked by reading, not compiled, not run on hardware
//
#include <hip/hip_runtime.h>
#include <math.h>

#define BSZ 128
#define MD 128
#define NS 32768
#define NF 128
#define NC 257
#define WIN 512
#define HOP 256
#define NI 4
#define KP 544   // 2*257 = 514 padded up to 17*32 for WMMA K-loop

typedef __attribute__((ext_vector_type(16))) __bf16 v16bf;
typedef __attribute__((ext_vector_type(8)))  __bf16 v8bf;
typedef __attribute__((ext_vector_type(8)))  float  v8f;
typedef __attribute__((ext_vector_type(4)))  unsigned v4u;
typedef __attribute__((ext_vector_type(8)))  unsigned v8u;

// ---------------------------------------------------------------- zero init
__global__ void zero_kernel(float* __restrict__ out, float* __restrict__ tsig,
                            unsigned* __restrict__ selmax) {
  long idx = (long)blockIdx.x * blockDim.x + threadIdx.x;
  long stride = (long)gridDim.x * blockDim.x;
  for (long i = idx; i < (long)BSZ * NS; i += stride) out[i] = 0.f;
  for (long i = idx; i < (long)BSZ * NI * NS; i += stride) tsig[i] = 0.f;
  if (idx == 0) *selmax = 0u;
}

// ------------------------------------------------- WMMA bf16 GEMM + bias/act
// C(MxN) = act(A(MxK) @ W(KxN) + bias), one wave per 16x16 tile.
// act: 0=none, 1=lrelu, 2=sigmoid, 3=sigmoid*0.25
__global__ __launch_bounds__(32) void gemm_wmma(
    const float* __restrict__ A, const float* __restrict__ W,
    const float* __restrict__ bias, float* __restrict__ C,
    int M, int N, int K, int act) {
  int tn = blockIdx.x, tm = blockIdx.y;
  int lane = threadIdx.x & 31;
  int half = lane >> 4;        // 0: lanes 0-15, 1: lanes 16-31
  int l16  = lane & 15;
  v8f acc = {0.f, 0.f, 0.f, 0.f, 0.f, 0.f, 0.f, 0.f};
  int m = tm * 16 + l16;                 // A row (M multiple of 16)
  int n = tn * 16 + l16;                 // C col
  int nc = n < N ? n : (N - 1);          // clamped col: OOB lanes read garbage
  const float* Wc = W + nc;              // that never reaches guarded stores
  for (int k0 = 0; k0 < K; k0 += 32) {
    // A 16x32 bf16 layout: per lane, k = k0+half*8+{0..7} and k0+16+half*8+{0..7}
    const float* arow = A + (long)m * K + k0 + half * 8;
    float4 a0 = *(const float4*)(arow);
    float4 a1 = *(const float4*)(arow + 4);
    float4 a2 = *(const float4*)(arow + 16);
    float4 a3 = *(const float4*)(arow + 20);
    v16bf a, b;
    a[0]=(__bf16)a0.x;  a[1]=(__bf16)a0.y;  a[2]=(__bf16)a0.z;  a[3]=(__bf16)a0.w;
    a[4]=(__bf16)a1.x;  a[5]=(__bf16)a1.y;  a[6]=(__bf16)a1.z;  a[7]=(__bf16)a1.w;
    a[8]=(__bf16)a2.x;  a[9]=(__bf16)a2.y;  a[10]=(__bf16)a2.z; a[11]=(__bf16)a2.w;
    a[12]=(__bf16)a3.x; a[13]=(__bf16)a3.y; a[14]=(__bf16)a3.z; a[15]=(__bf16)a3.w;
    int kb0 = k0 + half * 8;             // B 32x16 bf16 layout: k = kb0+{0..7}, kb0+16+{0..7}
#pragma unroll
    for (int e = 0; e < 16; ++e)
      b[e] = (__bf16)Wc[(long)(kb0 + e + (e >= 8 ? 8 : 0)) * N];
    acc = __builtin_amdgcn_wmma_f32_16x16x32_bf16(false, a, false, b,
                                                  (short)0, acc, false, false);
  }
  if (n < N) {
    float bv = bias[n];
#pragma unroll
    for (int r = 0; r < 8; ++r) {
      int row = tm * 16 + half * 8 + r;  // C/D layout: VGPR r -> M = r + half*8
      float v = acc[r] + bv;
      if (act == 1)      v = v > 0.f ? v : 0.2f * v;
      else if (act == 2) v = 1.f / (1.f + __expf(-v));
      else if (act == 3) v = 0.25f / (1.f + __expf(-v));
      C[(long)row * N + n] = v;
    }
  }
}

// ---------------------------------------------------- global max of sel pdf
__global__ void selmax_kernel(const float* __restrict__ means,
                              const float* __restrict__ stds,
                              unsigned* __restrict__ selmax) {
  int t = threadIdx.x;  // 512 = B*NI
  float m = means[t], sd = stds[t];
  float rng = rintf(m * 32767.f) / 32767.f;   // grid point nearest the mean
  float z = (rng - m) / sd;
  float p = __expf(-0.5f * z * z) / (sd * 2.5066282746f);
  atomicMax(selmax, __float_as_uint(p));      // p > 0: uint order == float order
}

// -------------------------------------------- ConvTranspose1d(k=4,s=2,p=1)
// Input tile staged into LDS by the Tensor Data Mover (TDM).
__global__ void convt_kernel(const float* __restrict__ h, const float* __restrict__ w,
                             const float* __restrict__ bias, float* __restrict__ y,
                             int Lin) {
  __shared__ float sh[MD * 64];
  int b = blockIdx.x;
  int Lout = Lin * 2;
  unsigned nelem = (unsigned)(MD * Lin);
  // ---- TDM descriptor (D#): 1-D tile of nelem f32, global -> LDS ----
  unsigned long long ga = (unsigned long long)(h + (long)b * MD * Lin);
  unsigned lds_off = (unsigned)(unsigned long long)(&sh[0]);
  v4u g0;
  g0[0] = 1u;                                          // count=1, user mode
  g0[1] = lds_off;                                     // lds_addr
  g0[2] = (unsigned)ga;                                // global_addr[31:0]
  g0[3] = (unsigned)((ga >> 32) & 0x1ffffffu) | (2u << 30);  // addr[56:32]|type=2
  v8u g1;
  g1[0] = (2u << 16);                                  // data_size = 4 bytes
  g1[1] = (nelem & 0xffffu) << 16;                     // tensor_dim0[15:0]
  g1[2] = (nelem >> 16) | (1u << 16);                  // dim0[31:16], tensor_dim1=1
  g1[3] = (nelem & 0xffffu) << 16;                     // tile_dim0 (<=8192)
  g1[4] = 0u;                                          // tile_dim1/2 unused
  g1[5] = nelem;                                       // tensor_dim0_stride
  g1[6] = 0u;
  g1[7] = 0u;
  if (threadIdx.x < 32) {
    asm volatile("tensor_load_to_lds %0, %1" :: "s"(g0), "s"(g1) : "memory");
    __builtin_amdgcn_s_wait_tensorcnt(0);
  }
  __syncthreads();

  for (int idx = threadIdx.x; idx < MD * Lout; idx += blockDim.x) {
    int co = idx / Lout, t = idx % Lout;
    int js[2], qs[2]; int nv = 0;
#pragma unroll
    for (int j = 0; j < 4; ++j) {          // y[co,t] = sum_ci,j k[co,ci,j]*xd[ci,t+j]
      int pm = t + j - 2;                  // lhs-dilation 2, pad 2
      if (pm >= 0 && !(pm & 1)) { int q = pm >> 1; if (q < Lin) { js[nv] = j; qs[nv] = q; ++nv; } }
    }
    float acc = bias[co];
    const float* wc = w + (long)co * MD * 4;
    if (nv == 2) {
      int j0 = js[0], q0 = qs[0], j1 = js[1], q1 = qs[1];
      for (int ci = 0; ci < MD; ++ci)
        acc += wc[ci * 4 + j0] * sh[ci * Lin + q0] + wc[ci * 4 + j1] * sh[ci * Lin + q1];
    } else {
      int j0 = js[0], q0 = qs[0];
      for (int ci = 0; ci < MD; ++ci)
        acc += wc[ci * 4 + j0] * sh[ci * Lin + q0];
    }
    y[(long)b * MD * Lout + idx] = acc;
  }
}

// --------------------------------------------- ExampleNorm + leaky relu
__global__ void norm_lrelu_kernel(float* __restrict__ y, int L) {
  __shared__ float ssum[256], ssq[256];
  int b = blockIdx.x; int n = MD * L;
  float s = 0.f, sq = 0.f;
  for (int i = threadIdx.x; i < n; i += blockDim.x) {
    float v = y[(long)b * n + i]; s += v; sq += v * v;
  }
  ssum[threadIdx.x] = s; ssq[threadIdx.x] = sq;
  __syncthreads();
  for (int off = blockDim.x / 2; off > 0; off >>= 1) {
    if ((int)threadIdx.x < off) {
      ssum[threadIdx.x] += ssum[threadIdx.x + off];
      ssq[threadIdx.x]  += ssq[threadIdx.x + off];
    }
    __syncthreads();
  }
  float mean = ssum[0] / n;
  float var  = ssq[0] / n - mean * mean;
  float inv  = 1.f / (sqrtf(fmaxf(var, 0.f)) + 1e-8f);
  for (int i = threadIdx.x; i < n; i += blockDim.x) {
    float v = y[(long)b * n + i] * inv;
    y[(long)b * n + i] = v > 0.f ? v : 0.2f * v;
  }
}

// --------------------------------- final Conv1d(128->1, k=7, p=3), squared
__global__ void env_final_kernel(const float* __restrict__ h, const float* __restrict__ wf,
                                 const float* __restrict__ bf, float* __restrict__ envS) {
  int b = blockIdx.x, t = threadIdx.x;  // 128 threads
  float acc = bf[0];
  for (int ci = 0; ci < MD; ++ci) {
    const float* hp = h + ((long)b * MD + ci) * NF;
    const float* wp = wf + ci * 7;
#pragma unroll
    for (int j = 0; j < 7; ++j) {
      int u = t + j - 3;
      if (u >= 0 && u < NF) acc += wp[j] * hp[u];
    }
  }
  envS[b * NF + t] = acc * acc;
}

// --------------------------- linear interp 128 -> 32768, multiply by noise
__global__ void interp_kernel(const float* __restrict__ envS, const float* __restrict__ noise,
                              float* __restrict__ env32) {
  int b = blockIdx.x;
  for (int s = threadIdx.x; s < NS; s += blockDim.x) {
    float c = (s + 0.5f) * (128.f / 32768.f) - 0.5f;
    c = fminf(fmaxf(c, 0.f), 127.f);
    int i0 = (int)c; int i1 = min(i0 + 1, 127);
    float w = c - (float)i0;
    float e = envS[b * NF + i0] * (1.f - w) + envS[b * NF + i1] * w;
    env32[(long)b * NS + s] = e * noise[s];
  }
}

// ---- build transposed DFT basis DmatT[n][k], bf16, window+ortho folded in ----
// k even = 2*kk:  g*cos(2*pi*kk*n/512) * win(n) / sqrt(512)
// k odd  = 2*kk+1: -g*sin(...) * win(n) / sqrt(512);  g = 1 for kk in {0,256} else 2
__global__ void init_dmat_kernel(__bf16* __restrict__ DT) {
  const float inv = 0.04419417382f;  // 1/sqrt(512)
  for (long idx = (long)blockIdx.x * blockDim.x + threadIdx.x;
       idx < (long)WIN * KP; idx += (long)gridDim.x * blockDim.x) {
    int n = (int)(idx / KP), k = (int)(idx % KP);
    float v = 0.f;
    if (k < 2 * NC) {
      int kk = k >> 1;
      float g = (kk == 0 || kk == NC - 1) ? 1.f : 2.f;
      float phi = 6.2831853072f * (float)((kk * n) & 511) / 512.f;
      float cs, sn; __sincosf(phi, &sn, &cs);
      float win = 0.54f - 0.46f * __cosf(6.2831853072f * (float)n / 512.f);
      v = ((k & 1) ? -g * sn : g * cs) * win * inv;
    }
    DT[idx] = (__bf16)v;
  }
}

// ---- per-(b,i): frames = X(128x514) @ D(514x512) on WMMA, OLA into tsig ----
// X row f holds limit_norm(c)^(f+1) interleaved re/im; built in 139KB LDS (bf16).
__global__ __launch_bounds__(256) void frame_dft_wmma_kernel(
    const float* __restrict__ tf, const __bf16* __restrict__ DT,
    float* __restrict__ tsig) {
  extern __shared__ __bf16 Ash[];      // [NF][KP] = 128*544 bf16 = 139264 B
  int bi = blockIdx.x;                 // b*4 + i
  const float* row = tf + (long)(bi >> 2) * 2056 + (bi & 3) * 514;
  // build A: coefficients c^(f+1) per (frame f, coeff k)
  for (int t = threadIdx.x; t < NF * NC; t += blockDim.x) {
    int f = t / NC, k = t % NC;
    float p = (float)(f + 1);
    float cre = row[2 * k], cim = row[2 * k + 1];
    float nrm = sqrtf(cre * cre + cim * cim);
    float sc  = fminf(nrm, 0.9999f) / (nrm + 1e-8f);   // limit_norm
    float mag = nrm * sc;
    float mp  = (mag > 0.f) ? __expf(p * __logf(mag)) : 0.f;
    float ang = p * atan2f(cim * sc, cre * sc);
    float ca, sa; __sincosf(ang, &sa, &ca);
    Ash[f * KP + 2 * k]     = (__bf16)(mp * ca);
    Ash[f * KP + 2 * k + 1] = (__bf16)(mp * sa);
  }
  for (int t = threadIdx.x; t < NF * (KP - 2 * NC); t += blockDim.x) {
    int f = t / (KP - 2 * NC), k = 2 * NC + t % (KP - 2 * NC);
    Ash[f * KP + k] = (__bf16)0.f;     // K padding
  }
  __syncthreads();
  // WMMA: 8x32 = 256 output tiles of 16x16, 8 waves round-robin
  int wid  = threadIdx.x >> 5;
  int lane = threadIdx.x & 31;
  int half = lane >> 4, l16 = lane & 15;
  for (int tile = wid; tile < 256; tile += 8) {
    int tm = tile >> 5, tn = tile & 31;
    int m = tm * 16 + l16;             // frame index for A row
    int n = tn * 16 + l16;             // sample index for B col
    v8f acc = {0.f, 0.f, 0.f, 0.f, 0.f, 0.f, 0.f, 0.f};
    for (int k0 = 0; k0 < KP; k0 += 32) {
      const v8bf* ap = (const v8bf*)(Ash + m * KP + k0 + half * 8);   // ds b128
      v8bf alo = ap[0], ahi = ap[2];   // +0 and +16 elements
      const v8bf* bp = (const v8bf*)(DT + (long)n * KP + k0 + half * 8);
      v8bf blo = bp[0], bhi = bp[2];   // +0 and +16 elements
      v16bf a, b;
#pragma unroll
      for (int e = 0; e < 8; ++e) {
        a[e] = alo[e]; a[e + 8] = ahi[e];
        b[e] = blo[e]; b[e + 8] = bhi[e];
      }
      acc = __builtin_amdgcn_wmma_f32_16x16x32_bf16(false, a, false, b,
                                                    (short)0, acc, false, false);
    }
    // epilogue: overlap-add t[f][n] (window already folded into D)
#pragma unroll
    for (int r = 0; r < 8; ++r) {
      int f = tm * 16 + half * 8 + r;
      int s = f * HOP + n;
      if (s < NS) atomicAdd(&tsig[(long)bi * NS + s], acc[r]);
    }
  }
}

// ---- 32768-pt spectral circular convolution, whole signal in 256KB LDS ----
__global__ __launch_bounds__(1024) void fft_conv_kernel(
    const float* __restrict__ env32, const float* __restrict__ tsig,
    const float* __restrict__ means, const float* __restrict__ stds,
    const unsigned* __restrict__ selmax, float* __restrict__ out) {
  extern __shared__ float smem[];
  float* re = smem;
  float* im = smem + NS;
  int bi = blockIdx.x;
  int b  = bi >> 2;
  int tid = threadIdx.x;
  float mean = means[bi], sd = stds[bi];
  float smax = __uint_as_float(*selmax) + 1e-8f;
  float invsd = 1.f / sd;
  float pk = 1.f / (sd * 2.5066282746f * smax);
  for (int n = tid; n < NS; n += 1024) {      // load with 15-bit bit-reversal
    float rng = (float)n * (1.f / 32767.f);
    float z = (rng - mean) * invsd;
    float a = env32[(long)b * NS + n] * __expf(-0.5f * z * z) * pk;
    float bb = tsig[(long)bi * NS + n];
    int r = (int)(__brev((unsigned)n) >> 17);
    re[r] = a; im[r] = bb;
  }
  __syncthreads();
  for (int s = 1; s <= 15; ++s) {             // forward FFT, e^{-i}
    int mh = 1 << (s - 1);
    float base = -3.14159265359f / (float)mh;
    for (int t = tid; t < NS / 2; t += 1024) {
      int j  = t & (mh - 1);
      int i0 = ((t >> (s - 1)) << s) + j;
      int i1 = i0 + mh;
      float ang = base * (float)j;
      float wsn, wcs; __sincosf(ang, &wsn, &wcs);
      float xr = re[i1], xi = im[i1];
      float tr = xr * wcs - xi * wsn, ti = xr * wsn + xi * wcs;
      float ur = re[i0], ui = im[i0];
      re[i0] = ur + tr; im[i0] = ui + ti;
      re[i1] = ur - tr; im[i1] = ui - ti;
    }
    __syncthreads();
  }
  // Hermitian split: A=(Z(k)+conj(Z(N-k)))/2, B=(Z(k)-conj(Z(N-k)))/(2i); D=A*B
  for (int k = tid; k < NS / 2; k += 1024) {
    if (k == 0) {
      re[0] = re[0] * im[0];                im[0] = 0.f;
      re[NS / 2] = re[NS / 2] * im[NS / 2]; im[NS / 2] = 0.f;
    } else {
      int km = NS - k;
      float zkr = re[k], zki = im[k], zmr = re[km], zmi = im[km];
      float Ar = 0.5f * (zkr + zmr), Ai = 0.5f * (zki - zmi);
      float Br = 0.5f * (zki + zmi), Bq = 0.5f * (zmr - zkr);
      float Dr = Ar * Br - Ai * Bq;
      float Di = Ar * Bq + Ai * Br;
      re[k] = Dr;  im[k] = Di;
      re[km] = Dr; im[km] = -Di;              // D Hermitian (both inputs real)
    }
  }
  __syncthreads();
  for (int n = tid; n < NS; n += 1024) {      // in-place bit-reverse permute
    int r = (int)(__brev((unsigned)n) >> 17);
    if (n < r) {
      float t1 = re[n]; re[n] = re[r]; re[r] = t1;
      float t2 = im[n]; im[n] = im[r]; im[r] = t2;
    }
  }
  __syncthreads();
  for (int s = 1; s <= 15; ++s) {             // inverse FFT, e^{+i}, unnormalized
    int mh = 1 << (s - 1);
    float base = 3.14159265359f / (float)mh;
    for (int t = tid; t < NS / 2; t += 1024) {
      int j  = t & (mh - 1);
      int i0 = ((t >> (s - 1)) << s) + j;
      int i1 = i0 + mh;
      float ang = base * (float)j;
      float wsn, wcs; __sincosf(ang, &wsn, &wcs);
      float xr = re[i1], xi = im[i1];
      float tr = xr * wcs - xi * wsn, ti = xr * wsn + xi * wcs;
      float ur = re[i0], ui = im[i0];
      re[i0] = ur + tr; im[i0] = ui + ti;
      re[i1] = ur - tr; im[i1] = ui - ti;
    }
    __syncthreads();
  }
  const float scale = 0.25f / (32768.f * 181.01933598f);  // ortho + mean over NI
  for (int n = tid; n < NS; n += 1024)
    atomicAdd(&out[(long)b * NS + n], re[n] * scale);
}

// --------------------------------------------------------------------------
extern "C" void kernel_launch(void* const* d_in, const int* in_sizes, int n_in,
                              void* d_out, int out_size, void* d_ws, size_t ws_size,
                              hipStream_t stream) {
  (void)in_sizes; (void)n_in; (void)out_size; (void)ws_size;
  const float* x      = (const float*)d_in[0];
  const float* noise  = (const float*)d_in[1];
  const float* env_w0 = (const float*)d_in[2];
  const float* env_b0 = (const float*)d_in[3];
  const float* env_ct[5]  = {(const float*)d_in[4], (const float*)d_in[5], (const float*)d_in[6],
                             (const float*)d_in[7], (const float*)d_in[8]};
  const float* env_ctb[5] = {(const float*)d_in[9], (const float*)d_in[10], (const float*)d_in[11],
                             (const float*)d_in[12], (const float*)d_in[13]};
  const float* env_wf = (const float*)d_in[14];
  const float* env_bf = (const float*)d_in[15];
  const float* tr_w[3]   = {(const float*)d_in[16], (const float*)d_in[17], (const float*)d_in[18]};
  const float* tr_b[3]   = {(const float*)d_in[19], (const float*)d_in[20], (const float*)d_in[21]};
  const float* mean_w[3] = {(const float*)d_in[22], (const float*)d_in[23], (const float*)d_in[24]};
  const float* mean_b[3] = {(const float*)d_in[25], (const float*)d_in[26], (const float*)d_in[27]};
  const float* std_w[3]  = {(const float*)d_in[28], (const float*)d_in[29], (const float*)d_in[30]};
  const float* std_b[3]  = {(const float*)d_in[31], (const float*)d_in[32], (const float*)d_in[33]};

  float* ws    = (float*)d_ws;
  float* t1    = ws;                       // 16384
  float* t2    = t1 + 16384;               // 16384
  float* means = t2 + 16384;               // 512
  float* stds  = means + 512;              // 512
  float* tf    = stds + 512;               // 263168
  float* e0    = tf + 263168;              // 65536
  float* cA    = e0 + 65536;               // 2097152
  float* cB    = cA + 2097152;             // 2097152
  float* envS  = cB + 2097152;             // 16384
  float* env32 = envS + 16384;             // 4194304
  float* tsig  = env32 + 4194304;          // 16777216
  float* tail  = tsig + 16777216;
  unsigned* selmax = (unsigned*)tail;
  __bf16* dmat = (__bf16*)(tail + 16);     // 544*512 bf16 = 557056 B
  float* out = (float*)d_out;

  hipFuncSetAttribute((const void*)fft_conv_kernel,
                      hipFuncAttributeMaxDynamicSharedMemorySize,
                      (int)(2 * NS * sizeof(float)));
  hipFuncSetAttribute((const void*)frame_dft_wmma_kernel,
                      hipFuncAttributeMaxDynamicSharedMemorySize,
                      (int)(NF * KP * sizeof(__bf16)));

  zero_kernel<<<2048, 256, 0, stream>>>(out, tsig, selmax);
  init_dmat_kernel<<<1024, 256, 0, stream>>>(dmat);

  auto gemm = [&](const float* A, const float* W, const float* bias, float* C,
                  int M, int N, int K, int act) {
    dim3 g((N + 15) / 16, M / 16);
    gemm_wmma<<<g, 32, 0, stream>>>(A, W, bias, C, M, N, K, act);
  };
  // mean stack
  gemm(x,  mean_w[0], mean_b[0], t1,    BSZ, MD, MD, 1);
  gemm(t1, mean_w[1], mean_b[1], t2,    BSZ, MD, MD, 1);
  gemm(t2, mean_w[2], mean_b[2], means, BSZ, NI, MD, 2);
  // std stack
  gemm(x,  std_w[0], std_b[0], t1,   BSZ, MD, MD, 1);
  gemm(t1, std_w[1], std_b[1], t2,   BSZ, MD, MD, 1);
  gemm(t2, std_w[2], std_b[2], stds, BSZ, NI, MD, 3);
  // transfer-function stack
  gemm(x,  tr_w[0], tr_b[0], t1, BSZ, MD,   MD, 1);
  gemm(t1, tr_w[1], tr_b[1], t2, BSZ, MD,   MD, 1);
  gemm(t2, tr_w[2], tr_b[2], tf, BSZ, 2056, MD, 0);
  // envelope first linear (B,128)->(B,512)
  gemm(x, env_w0, env_b0, e0, BSZ, 512, MD, 0);

  selmax_kernel<<<1, 512, 0, stream>>>(means, stds, selmax);

  // 5x ConvTranspose (TDM-staged input) + ExampleNorm + lrelu (ping-pong)
  const float* cur = e0;
  float* bufs[2] = {cA, cB};
  int Lin = 4;
  for (int l = 0; l < 5; ++l) {
    float* dst = bufs[l & 1];
    convt_kernel<<<BSZ, 256, 0, stream>>>(cur, env_ct[l], env_ctb[l], dst, Lin);
    norm_lrelu_kernel<<<BSZ, 256, 0, stream>>>(dst, Lin * 2);
    cur = dst; Lin *= 2;
  }
  env_final_kernel<<<BSZ, 128, 0, stream>>>(cur, env_wf, env_bf, envS);
  interp_kernel<<<BSZ, 256, 0, stream>>>(envS, noise, env32);

  // per-frame irfft + window + overlap-add as one big WMMA GEMM per (b,i)
  frame_dft_wmma_kernel<<<BSZ * NI, 256, NF * KP * sizeof(__bf16), stream>>>(
      tf, dmat, tsig);

  fft_conv_kernel<<<BSZ * NI, 1024, 2 * NS * sizeof(float), stream>>>(
      env32, tsig, means, stds, selmax, out);
}
